// DynamicModel_12000138625657
// MI455X (gfx1250) — compile-verified, matrix operand order checked by reading, not run on hardware
//
#include <hip/hip_runtime.h>
#include <hip/hip_bf16.h>

typedef __attribute__((ext_vector_type(16))) __bf16 v16bf;
typedef __attribute__((ext_vector_type(8)))  float  v8f;

#define NT 4000
#define NN 64
#define TCUR 3999           // valid "cur" time indices: 0..3998
#define DT 0.01f
#define ONE_MINUS_DT 0.99f

union BF16x16 {
  __bf16       h[16];
  unsigned int u[8];
  v16bf        v;
};

// Preferred: native f32->bf16 converts (clang legalizes; can fuse adjacent
// pairs into v_cvt_pk_bf16_f32 on gfx1250). Element 2j -> bits [15:0].
__device__ __forceinline__ void put_bf16_pair(BF16x16& f, int j, float lo, float hi) {
  f.h[2 * j]     = (__bf16)lo;
  f.h[2 * j + 1] = (__bf16)hi;
}

__device__ __forceinline__ float fast_tanh(float x) {
#if __has_builtin(__builtin_amdgcn_tanhf)
  return __builtin_amdgcn_tanhf(x);   // gfx1250 v_tanh_f32 (trans pipe, co-executes with WMMA)
#else
  return tanhf(x);
#endif
}

// Per-lane K base for 16-bit A/B fragments (ISA 7.12.2, 16-bit A 16x32 layout):
// VGPR j holds K = (j<4 ? 0 : 16) + half*8 + 2*(j&3), K+1
__device__ __forceinline__ int frag_kbase(int j, int half) {
  return ((j < 4) ? 0 : 16) + half * 8 + 2 * (j & 3);
}

__global__ __launch_bounds__(256) void rnn_step_wmma(
    const float* __restrict__ X,   // trajectory [512][64][4000]
    const float* __restrict__ W,   // weight [64][64]
    float* __restrict__ out)       // [512][64][4000]
{
  const int lane = threadIdx.x & 31;
  const int half = lane >> 4;       // lane group 0..15 vs 16..31
  const int r    = lane & 15;

  const int wave  = blockIdx.x * 8 + (threadIdx.x >> 5);  // 12800 waves total
  const int b     = wave / 25;                            // batch 0..511
  const int strip = wave % 25;                            // 25 strips x 160 t each
  const long base = (long)b * NN * NT;

  // ---- Load A fragments (weights) once; reused for all 10 time tiles ----
  // aF[mt][ks]: 16(m) x 32(k) bf16 tile, m-tile mt, K-step ks (n = ks*32 + k)
  BF16x16 aF[4][2];
#pragma unroll
  for (int mt = 0; mt < 4; ++mt) {
    const int m = mt * 16 + r;
#pragma unroll
    for (int ks = 0; ks < 2; ++ks) {
#pragma unroll
      for (int j = 0; j < 8; ++j) {
        const int k = ks * 32 + frag_kbase(j, half);
        put_bf16_pair(aF[mt][ks], j, W[m * 64 + k], W[m * 64 + k + 1]);
      }
    }
  }

#pragma unroll 1
  for (int tile = 0; tile < 10; ++tile) {
    const int t  = strip * 160 + tile * 16 + r;   // input time column this lane handles
    const bool tv = (t < TCUR);

    // ---- Build B fragments: 32(n) x 16(t) of tanh(X), two K-steps ----
    BF16x16 b0, b1;
#pragma unroll
    for (int j = 0; j < 8; ++j) {
      const int k = frag_kbase(j, half);
      float x0 = tv ? X[base + (long)(k)      * NT + t] : 0.0f;
      float x1 = tv ? X[base + (long)(k + 1)  * NT + t] : 0.0f;
      put_bf16_pair(b0, j, fast_tanh(x0), fast_tanh(x1));
      float y0 = tv ? X[base + (long)(k + 32) * NT + t] : 0.0f;
      float y1 = tv ? X[base + (long)(k + 33) * NT + t] : 0.0f;
      put_bf16_pair(b1, j, fast_tanh(y0), fast_tanh(y1));
    }

    // ---- 4 m-tiles of 16x16 output; K=64 via two chained WMMAs ----
    // WMMAs are unconditional (EXEC must be all-ones); only loads/stores predicated.
#pragma unroll
    for (int mt = 0; mt < 4; ++mt) {
      v8f acc = {};
      acc = __builtin_amdgcn_wmma_f32_16x16x32_bf16(
          false, aF[mt][0].v, false, b0.v, (short)0, acc, false, false);
      acc = __builtin_amdgcn_wmma_f32_16x16x32_bf16(
          false, aF[mt][1].v, false, b1.v, (short)0, acc, false, false);
      if (tv) {
#pragma unroll
        for (int v = 0; v < 8; ++v) {
          const int m = mt * 16 + v + 8 * half;       // C/D layout: M = v + 8*half
          const float c = X[base + (long)m * NT + t]; // L0/L2 hit: same tile as B loads
          out[base + (long)m * NT + t + 1] = ONE_MINUS_DT * c + DT * acc[v];
        }
      }
    }
  }
}

__global__ void copy_t0(const float* __restrict__ X, float* __restrict__ out) {
  int i = blockIdx.x * blockDim.x + threadIdx.x;   // over 512*64 (b,m) pairs
  if (i < 512 * 64) out[(long)i * NT] = X[(long)i * NT];
}

extern "C" void kernel_launch(void* const* d_in, const int* in_sizes, int n_in,
                              void* d_out, int out_size, void* d_ws, size_t ws_size,
                              hipStream_t stream) {
  const float* X = (const float*)d_in[0];   // trajectory (512,64,4000) f32
  const float* W = (const float*)d_in[1];   // weight (64,64) f32
  float* out = (float*)d_out;

  // 12800 waves / 8 waves per 256-thread block = 1600 blocks
  rnn_step_wmma<<<dim3(1600), dim3(256), 0, stream>>>(X, W, out);
  copy_t0<<<dim3((512 * 64 + 255) / 256), dim3(256), 0, stream>>>(X, out);
}
